// ReducedRACE_65618510348698
// MI455X (gfx1250) — matrix operand order repaired and checked by reading, not compiled, not run on hardware
//
#include <hip/hip_runtime.h>
#include <hip/hip_bf16.h>
#include <math.h>

// ---------------- problem constants (from reference) ----------------
#define N_NODES 10000
#define G_GRAPHS 100
#define E_EDGES 400000
#define C_CH 32
#define LM 16
#define S_SP 4
#define NB 8
#define NLAY 3
#define CUTR 6.0f
#define AVGN 40.0f

typedef __attribute__((ext_vector_type(16))) _Float16 v16h;
typedef __attribute__((ext_vector_type(8)))  float    v8f;
typedef __attribute__((ext_vector_type(2)))  float    v2f;

__device__ __forceinline__ int lof_of(int mm) {
    return (mm == 0) ? 0 : (mm < 4) ? 1 : (mm < 9) ? 2 : 3;
}
__device__ __forceinline__ float fast_rcp(float x) { return __builtin_amdgcn_rcpf(x); }
__device__ __forceinline__ float silu(float x) {
    return x * fast_rcp(1.0f + __expf(-x));     // v_rcp_f32, not div sequence
}
__device__ __forceinline__ void atomic_add_f32(float* p, float v) {
    __hip_atomic_fetch_add(p, v, __ATOMIC_RELAXED, __HIP_MEMORY_SCOPE_AGENT);
}

// =====================================================================
// Kernel 1: per-edge geometry -> Y (16 sph-harm, f32) and ef (8 bessel, f16)
// =====================================================================
__global__ void edge_geom_kernel(const float* __restrict__ pos,
                                 const float* __restrict__ cell,
                                 const int*   __restrict__ Sij,
                                 const int*   __restrict__ eidx,
                                 const int*   __restrict__ batch,
                                 float* __restrict__ Yout,
                                 _Float16* __restrict__ efh)
{
    int e = blockIdx.x * blockDim.x + threadIdx.x;
    if (e >= E_EDGES) return;
    int ia = eidx[e];
    int ja = eidx[E_EDGES + e];
    int g  = batch[ia];
    float sx = (float)Sij[e*3+0], sy = (float)Sij[e*3+1], sz = (float)Sij[e*3+2];
    const float* cg = cell + (size_t)g * 9;
    float shx = sx*cg[0] + sy*cg[3] + sz*cg[6];
    float shy = sx*cg[1] + sy*cg[4] + sz*cg[7];
    float shz = sx*cg[2] + sy*cg[5] + sz*cg[8];
    float rx = (pos[ja*3+0] - pos[ia*3+0] + shx) * (1.0f/CUTR);
    float ry = (pos[ja*3+1] - pos[ia*3+1] + shy) * (1.0f/CUTR);
    float rz = (pos[ja*3+2] - pos[ia*3+2] + shz) * (1.0f/CUTR);
    float r  = sqrtf(rx*rx + ry*ry + rz*rz);
    float rs = (r > 1e-9f) ? r : 1e-9f;
    float inv = fast_rcp(rs);
    float x = rx*inv, y = ry*inv, z = rz*inv;

    const float s3  = 1.7320508075688772f;
    const float s5  = 2.2360679774997896f;
    const float s15 = 3.8729833462074170f;
    const float s7  = 2.6457513110645906f;
    const float s105 = 10.246950765959598f;
    const float s35_8 = 2.0916500663351889f;   // sqrt(35/8)
    const float s21_8 = 1.6201851746019651f;   // sqrt(21/8)

    float* Yp = Yout + (size_t)e * LM;
    Yp[0]  = 1.0f;
    Yp[1]  = s3 * x;
    Yp[2]  = s3 * y;
    Yp[3]  = s3 * z;
    Yp[4]  = s15 * x * y;
    Yp[5]  = s15 * y * z;
    Yp[6]  = 0.5f * s5 * (3.0f*z*z - 1.0f);
    Yp[7]  = s15 * x * z;
    Yp[8]  = 0.5f * s15 * (x*x - y*y);
    Yp[9]  = s35_8 * y * (3.0f*x*x - y*y);
    Yp[10] = s105 * x * y * z;
    Yp[11] = s21_8 * y * (5.0f*z*z - 1.0f);
    Yp[12] = 0.5f * s7 * (5.0f*z*z*z - 3.0f*z);
    Yp[13] = s21_8 * x * (5.0f*z*z - 1.0f);
    Yp[14] = 0.5f * s105 * z * (x*x - y*y);
    Yp[15] = s35_8 * x * (x*x - 3.0f*y*y);

    // bessel * polynomial cutoff
    float fcut = (r < 1.0f) ? (1.0f - 6.0f*r*r + 8.0f*r*r*r - 3.0f*r*r*r*r) : 0.0f;
    const float PI = 3.14159265358979323846f;
    const float SQ2 = 1.4142135623730951f;
    _Float16* ep = efh + (size_t)e * NB;
    #pragma unroll
    for (int nb = 1; nb <= NB; nb++) {
        float b = SQ2 * __sinf(PI * (float)nb * rs) * inv;
        ep[nb-1] = (_Float16)(b * fcut);
    }
}

// =====================================================================
// Kernel 2: node embedding: h0, xnode, h init (layout h[n][c][mm])
// =====================================================================
__global__ void node_init_kernel(const int* __restrict__ species,
                                 const float* __restrict__ W_emb,
                                 const float* __restrict__ W_xtp,
                                 float* __restrict__ h,
                                 float* __restrict__ xnode)
{
    int n = blockIdx.x * blockDim.x + threadIdx.x;
    if (n >= N_NODES) return;
    int sp = species[n];
    float xn = 0.0f;
    float* hp = h + (size_t)n * (C_CH * LM);
    #pragma unroll
    for (int c = 0; c < C_CH; c++) {
        float h0c = W_emb[sp * C_CH + c];
        hp[c*LM + 0] = h0c;
        #pragma unroll
        for (int mm = 1; mm < LM; mm++) hp[c*LM + mm] = 0.0f;
        xn += h0c * W_xtp[c * S_SP + sp];
    }
    xnode[n] = xn;
}

// =====================================================================
// Kernel 3: equivariant tensor-product matmul via f32 WMMA (full precision)
//   out[n][mm][d] = sum_c in[n][c*strideC + mm*strideM] * W4[lof(mm)][c][d]
// One wave = 16 nodes; 5 waves/block, 125 blocks = 625 exact tiles.
//   hu:   in = h   (strideC=LM, strideM=1)
//   hmsg: in = agg (strideC=1,  strideM=C_CH)
// =====================================================================
__launch_bounds__(160)
__global__ void tp_matmul_kernel(const float* __restrict__ in,
                                 const float* __restrict__ W4,   // (4,C,C)
                                 float* __restrict__ outp,       // [n][mm][d]
                                 int strideC, int strideM)
{
    const int wave = threadIdx.x >> 5;
    const int lid  = threadIdx.x & 31;
    const int lo   = lid & 15;
    const int hi   = lid >> 4;
    const int tile = blockIdx.x * 5 + wave;
    const int nbase = tile * 16;

    // A-fragment rows: lane handles node (nbase+lo); lanes 0-15 K=0,1; 16-31 K=2,3
    const float* inRow = in + (size_t)(nbase + lo) * (C_CH * LM);

    for (int mm = 0; mm < LM; mm++) {
        const float* W = W4 + (size_t)lof_of(mm) * C_CH * C_CH;
        #pragma unroll
        for (int nt = 0; nt < 2; nt++) {
            v8f acc = {};
            #pragma unroll
            for (int kc = 0; kc < 8; kc++) {      // K = 32 in chunks of 4
                int c0 = kc*4 + hi*2;
                v2f a, b;
                a[0] = inRow[(c0    ) * strideC + mm * strideM];
                a[1] = inRow[(c0 + 1) * strideC + mm * strideM];
                b[0] = W[(c0    ) * C_CH + nt*16 + lo];
                b[1] = W[(c0 + 1) * C_CH + nt*16 + lo];
                acc = __builtin_amdgcn_wmma_f32_16x16x4_f32(false, a, false, b,
                                                            (short)0, acc, false, false);
            }
            float* op = outp + (size_t)nbase * (LM*C_CH) + mm*C_CH + nt*16 + lo;
            #pragma unroll
            for (int v = 0; v < 8; v++)
                op[(size_t)(v + hi*8) * (LM*C_CH)] = acc[v];
        }
    }
}

// =====================================================================
// WMMA f16 fragment loaders (16-bit A/B layouts, §7.12.2; wave32)
// =====================================================================
__device__ __forceinline__ v16h load_a_frag(const _Float16* __restrict__ rowBase,
                                            int kbase, int hi)
{
    v16h a;
    #pragma unroll
    for (int i = 0; i < 16; i++) {
        int v = i >> 1, hb = i & 1;
        int k = ((v < 4) ? (2*v + hb) : (16 + 2*(v-4) + hb)) + hi*8;
        a[i] = rowBase[kbase + k];
    }
    return a;
}

__device__ __forceinline__ v16h load_b_frag(const _Float16* __restrict__ W,
                                            int ncols, int kbase, int col, int hi)
{
    v16h b;
    #pragma unroll
    for (int i = 0; i < 16; i++) {
        int k = kbase + ((i < 8) ? i : (8 + i)) + hi*8;
        b[i] = W[k*ncols + col];
    }
    return b;
}

// =====================================================================
// Kernel 4: edge MLP (WMMA f16->f32) + message formation + scatter-add
// One wave = 16 edges. Block = 4 waves = 64 edges. Grid = E/64 (exact).
// =====================================================================
__launch_bounds__(128)
__global__ void edge_mlp_kernel(const _Float16* __restrict__ efh,
                                const float* __restrict__ Yv,
                                const _Float16* __restrict__ W1h, // 8x64
                                const float* __restrict__ bb1,    // 64
                                const _Float16* __restrict__ W2h, // 64x64
                                const float* __restrict__ bb2,    // 64
                                const _Float16* __restrict__ W3h, // 64x256
                                const float* __restrict__ hu,
                                const int* __restrict__ eidx,
                                float* __restrict__ agg)
{
    __shared__ _Float16 act[4][16 * 64];   // per-wave activation staging
    const int wave = threadIdx.x >> 5;
    const int lid  = threadIdx.x & 31;
    const int lo   = lid & 15;
    const int hi   = lid >> 4;
    const int tile = blockIdx.x * 4 + wave;
    const int ebase = tile * 16;
    _Float16* myact = act[wave];

    // ---- layer 1: (16x8 ef) x (8x64 W1) ----
    v16h a1;
    #pragma unroll
    for (int i = 0; i < 16; i++) a1[i] = (_Float16)0;
    if (hi == 0) {
        const _Float16* ep = efh + (size_t)(ebase + lo) * NB;
        #pragma unroll
        for (int i = 0; i < NB; i++) a1[i] = ep[i];   // A lane K=0..7
    }
    #pragma unroll
    for (int nt = 0; nt < 4; nt++) {
        v16h b;
        #pragma unroll
        for (int i = 0; i < 16; i++) b[i] = (_Float16)0;
        if (hi == 0) {
            #pragma unroll
            for (int i = 0; i < NB; i++) b[i] = W1h[i*64 + nt*16 + lo];
        }
        v8f c = {};
        c = __builtin_amdgcn_wmma_f32_16x16x32_f16(false, a1, false, b,
                                                   (short)0, c, false, false);
        float bias = bb1[nt*16 + lo];
        #pragma unroll
        for (int v = 0; v < 8; v++)
            myact[(v + hi*8)*64 + nt*16 + lo] = (_Float16)silu(c[v] + bias);
    }
    __syncthreads();

    // ---- layer 2: (16x64) x (64x64 W2) ----
    v16h a2[2];
    #pragma unroll
    for (int kb = 0; kb < 2; kb++) a2[kb] = load_a_frag(myact + lo*64, kb*32, hi);
    float h2[4][8];
    #pragma unroll
    for (int nt = 0; nt < 4; nt++) {
        v8f c = {};
        #pragma unroll
        for (int kb = 0; kb < 2; kb++) {
            v16h b = load_b_frag(W2h, 64, kb*32, nt*16 + lo, hi);
            c = __builtin_amdgcn_wmma_f32_16x16x32_f16(false, a2[kb], false, b,
                                                       (short)0, c, false, false);
        }
        float bias = bb2[nt*16 + lo];
        #pragma unroll
        for (int v = 0; v < 8; v++) h2[nt][v] = silu(c[v] + bias);
    }
    __syncthreads();     // everyone done reading act (layer-1 result)
    #pragma unroll
    for (int nt = 0; nt < 4; nt++)
        #pragma unroll
        for (int v = 0; v < 8; v++)
            myact[(v + hi*8)*64 + nt*16 + lo] = (_Float16)h2[nt][v];
    __syncthreads();
    #pragma unroll
    for (int kb = 0; kb < 2; kb++) a2[kb] = load_a_frag(myact + lo*64, kb*32, hi);

    // per-row edge indices for this lane's 8 output rows
    int iarr[8], jarr[8];
    #pragma unroll
    for (int v = 0; v < 8; v++) {
        int e = ebase + v + hi*8;
        iarr[v] = eidx[e];
        jarr[v] = eidx[E_EDGES + e];
    }

    // ---- layer 3: (16x64) x (64x256 W3), paired p=0 / p=1 column tiles ----
    for (int nt = 0; nt < 8; nt++) {
        v8f c0 = {};
        v8f c1 = {};
        #pragma unroll
        for (int kb = 0; kb < 2; kb++) {
            v16h b0 = load_b_frag(W3h, 256, kb*32, nt*16 + lo, hi);        // p=0 half
            v16h b1 = load_b_frag(W3h, 256, kb*32, 128 + nt*16 + lo, hi);  // p=1 half
            c0 = __builtin_amdgcn_wmma_f32_16x16x32_f16(false, a2[kb], false, b0,
                                                        (short)0, c0, false, false);
            c1 = __builtin_amdgcn_wmma_f32_16x16x32_f16(false, a2[kb], false, b1,
                                                        (short)0, c1, false, false);
        }
        // this lane owns column col = nt*16+lo => channel ch, degree l
        int col = nt*16 + lo;
        int ch = col >> 2;
        int l  = col & 3;
        int mstart = (l == 0) ? 0 : (l == 1) ? 1 : (l == 2) ? 4 : 9;
        int mcount = (l == 0) ? 1 : (l == 1) ? 3 : (l == 2) ? 5 : 7;
        #pragma unroll
        for (int v = 0; v < 8; v++) {
            int e = ebase + v + hi*8;
            float w0 = c0[v], w1 = c1[v];
            const float* hup  = hu + (size_t)jarr[v] * (LM * C_CH);
            float*       aggp = agg + (size_t)iarr[v] * (LM * C_CH);
            const float* Yp   = Yv + (size_t)e * LM;
            float hj0 = hup[ch];   // mm = 0
            for (int t = 0; t < mcount; t++) {
                int mm = mstart + t;
                float contrib = w0 * Yp[mm] * hj0 + w1 * hup[mm*C_CH + ch];
                atomic_add_f32(&aggp[mm*C_CH + ch], contrib);
            }
        }
    }
}

// =====================================================================
// Kernel 5: node update: combine precomputed hmsg with self-connection
// =====================================================================
__global__ void node_update_kernel(const float* __restrict__ h,
                                   const float* __restrict__ hmsg,  // [n][mm][d]
                                   const float* __restrict__ xnode,
                                   const int*   __restrict__ species,
                                   const float* __restrict__ Wsc,   // (4,C,S,C)
                                   float* __restrict__ hnext)
{
    int idx = blockIdx.x * blockDim.x + threadIdx.x;
    if (idx >= N_NODES * C_CH) return;
    int n = idx >> 5, d = idx & 31;
    int sp = species[n];
    float xn = xnode[n] * (1.0f / AVGN);
    const float* hp = h    + (size_t)n * (C_CH * LM);
    const float* mp = hmsg + (size_t)n * (LM * C_CH);
    float* op = hnext + (size_t)n * (C_CH * LM);
    #pragma unroll
    for (int mm = 0; mm < LM; mm++) {
        int l = lof_of(mm);
        const float* Ws = Wsc + (size_t)l * C_CH * S_SP * C_CH;
        float sc = 0.0f;
        #pragma unroll
        for (int c = 0; c < C_CH; c++)
            sc += hp[c*LM + mm] * Ws[c*S_SP*C_CH + sp*C_CH + d];
        op[d*LM + mm] = mp[mm*C_CH + d] * xn + sc;
    }
}

// =====================================================================
// Kernel 6: per-layer scalar readout
// =====================================================================
__global__ void readout_kernel(const float* __restrict__ h,
                               const float* __restrict__ Wr1, // C x 64
                               const float* __restrict__ br1, // 64
                               const float* __restrict__ Wr2, // 64
                               float* __restrict__ outs, int layer)
{
    int n = blockIdx.x * blockDim.x + threadIdx.x;
    if (n >= N_NODES) return;
    const float* hp = h + (size_t)n * (C_CH * LM);   // h[:,c,0] at stride LM
    float e = 0.0f;
    for (int j = 0; j < 64; j++) {
        float t = br1[j];
        #pragma unroll
        for (int c = 0; c < C_CH; c++) t += hp[c*LM] * Wr1[c*64 + j];
        e += silu(t) * Wr2[j];
    }
    outs[n*NLAY + layer] = e;
}

// =====================================================================
// Kernel 7: final silu-sum + segment-sum per graph
// =====================================================================
__global__ void finalize_kernel(const float* __restrict__ outs,
                                const int* __restrict__ batch,
                                float* __restrict__ out)
{
    int n = blockIdx.x * blockDim.x + threadIdx.x;
    if (n >= N_NODES) return;
    float s = 0.0f;
    #pragma unroll
    for (int l = 0; l < NLAY; l++) s += silu(outs[n*NLAY + l]);
    atomic_add_f32(&out[batch[n]], s);
}

// =====================================================================
// Kernel 8: f32 -> f16 weight conversion (grid-stride)
// =====================================================================
__global__ void cvt_f16_kernel(const float* __restrict__ src,
                               _Float16* __restrict__ dst, int n)
{
    for (int i = blockIdx.x * blockDim.x + threadIdx.x; i < n;
         i += gridDim.x * blockDim.x)
        dst[i] = (_Float16)src[i];
}

// =====================================================================
// Host orchestration
// =====================================================================
extern "C" void kernel_launch(void* const* d_in, const int* in_sizes, int n_in,
                              void* d_out, int out_size, void* d_ws, size_t ws_size,
                              hipStream_t stream)
{
    const float* positions = (const float*)d_in[0];
    const float* cell      = (const float*)d_in[1];
    const int*   Sij       = (const int*)  d_in[2];
    const int*   eidx      = (const int*)  d_in[3];
    const int*   species   = (const int*)  d_in[4];
    const int*   batch     = (const int*)  d_in[5];
    const float* W_emb     = (const float*)d_in[6];
    const float* W_xtp     = (const float*)d_in[7];
    const float* W_up      = (const float*)d_in[8];
    const float* W1        = (const float*)d_in[9];
    const float* b1        = (const float*)d_in[10];
    const float* W2        = (const float*)d_in[11];
    const float* b2        = (const float*)d_in[12];
    const float* W3        = (const float*)d_in[13];
    const float* W_down    = (const float*)d_in[14];
    const float* W_sc      = (const float*)d_in[15];
    const float* Wr1       = (const float*)d_in[16];
    const float* br1       = (const float*)d_in[17];
    const float* Wr2       = (const float*)d_in[18];
    float* out = (float*)d_out;

    // ---- carve workspace ----
    char* base = (char*)d_ws;
    size_t off = 0;
    auto carve = [&](size_t bytes) -> char* {
        char* p = base + off;
        off = (off + bytes + 255) & ~(size_t)255;
        return p;
    };
    float*    Yv    = (float*)   carve((size_t)E_EDGES * LM * 4);
    _Float16* efh   = (_Float16*)carve((size_t)E_EDGES * NB * 2);
    float*    h_a   = (float*)   carve((size_t)N_NODES * C_CH * LM * 4);
    float*    h_b   = (float*)   carve((size_t)N_NODES * C_CH * LM * 4);
    float*    hu    = (float*)   carve((size_t)N_NODES * LM * C_CH * 4); // also reused as hmsg
    float*    agg   = (float*)   carve((size_t)N_NODES * LM * C_CH * 4);
    float*    xnode = (float*)   carve((size_t)N_NODES * 4);
    float*    outs  = (float*)   carve((size_t)N_NODES * NLAY * 4);
    _Float16* W1h   = (_Float16*)carve((size_t)NLAY * NB * 64 * 2);
    _Float16* W2h   = (_Float16*)carve((size_t)NLAY * 64 * 64 * 2);
    _Float16* W3h   = (_Float16*)carve((size_t)NLAY * 64 * 256 * 2);
    (void)ws_size; (void)n_in; (void)in_sizes; (void)out_size;

    hipMemsetAsync(out, 0, G_GRAPHS * sizeof(float), stream);

    cvt_f16_kernel<<<16, 256, 0, stream>>>(W1, W1h, NLAY * NB * 64);
    cvt_f16_kernel<<<48, 256, 0, stream>>>(W2, W2h, NLAY * 64 * 64);
    cvt_f16_kernel<<<192, 256, 0, stream>>>(W3, W3h, NLAY * 64 * 256);

    edge_geom_kernel<<<(E_EDGES + 255) / 256, 256, 0, stream>>>(
        positions, cell, Sij, eidx, batch, Yv, efh);
    node_init_kernel<<<(N_NODES + 255) / 256, 256, 0, stream>>>(
        species, W_emb, W_xtp, h_a, xnode);

    const float* hcur = h_a;
    float* hnxt = h_b;
    for (int layer = 0; layer < NLAY; layer++) {
        // hu[n][mm][d] = sum_c h[n][c][mm] * W_up[l][c][d]   (f32 WMMA)
        tp_matmul_kernel<<<125, 160, 0, stream>>>(
            hcur, W_up + (size_t)layer * 4 * C_CH * C_CH, hu,
            /*strideC=*/LM, /*strideM=*/1);
        hipMemsetAsync(agg, 0, (size_t)N_NODES * LM * C_CH * 4, stream);
        edge_mlp_kernel<<<E_EDGES / 64, 128, 0, stream>>>(
            efh, Yv,
            W1h + (size_t)layer * NB * 64,  b1 + (size_t)layer * 64,
            W2h + (size_t)layer * 64 * 64,  b2 + (size_t)layer * 64,
            W3h + (size_t)layer * 64 * 256,
            hu, eidx, agg);
        // hmsg[n][mm][d] = sum_c agg[n][mm][c] * W_down[l][c][d]  (f32 WMMA)
        // (hu buffer is dead after edge_mlp; reuse it for hmsg)
        tp_matmul_kernel<<<125, 160, 0, stream>>>(
            agg, W_down + (size_t)layer * 4 * C_CH * C_CH, hu,
            /*strideC=*/1, /*strideM=*/C_CH);
        node_update_kernel<<<(N_NODES * C_CH + 255) / 256, 256, 0, stream>>>(
            hcur, hu, xnode, species,
            W_sc + (size_t)layer * 4 * C_CH * S_SP * C_CH,
            hnxt);
        readout_kernel<<<(N_NODES + 255) / 256, 256, 0, stream>>>(
            hnxt, Wr1 + (size_t)layer * C_CH * 64, br1 + (size_t)layer * 64,
            Wr2 + (size_t)layer * 64, outs, layer);
        const float* t = hcur; hcur = hnxt; hnxt = (float*)t;
    }
    finalize_kernel<<<(N_NODES + 255) / 256, 256, 0, stream>>>(outs, batch, out);
}
